// WaveletProcessor_61443802136705
// MI455X (gfx1250) — matrix operand order repaired and checked by reading, not compiled, not run on hardware
//
#include <hip/hip_runtime.h>
#include <hip/hip_bf16.h>
#include <math.h>

// ---------------------------------------------------------------------------
// WaveletProcessor fused kernel for MI455X (gfx1250, wave32, WMMA).
//
//  * Pre-kernels convert conv weights fp32 -> f16 into d_ws as 24 slices of
//    [64 out][64 in] halves (taps: trend@0..2, d1@3..5, d2@6..10, d3@11..19;
//    1x1 convs @20..23). The softmax(rec_weights) blend is LINEAR in the
//    second conv, so sw_i is folded into the 1x1 weight slices at conversion
//    time and the four b2 biases are pre-blended into one 64-float vector.
//  * Main kernel: block = (batch, 256-position) tile, 8 waves. x tile
//    (264 pos incl. +-4 halo, zero-padded) converted to f16 in LDS as
//    [pos][ch] with 144B-stride bank skew. Each wave owns 32 positions (two
//    N=16 tiles) and all 64 output channels (4 M-tiles); A-fragments are
//    reused across both N-tiles in registers. ReLU'd hidden activations
//    stage through per-wave LDS to re-layout for the 1x1-conv B operand,
//    whose WMMAs accumulate straight into the single persistent blended
//    accumulator. Next branch's weights are prefetched under the current
//    branch's WMMAs.
//  * launch_bounds(256, 2): a 512-VGPR budget so the scheduler can pipeline
//    fragment loads in registers (rounds with a 256-VGPR cap spilled
//    fragments to scratch on the WMMA critical path — strictly worse than
//    running 2 waves/SIMD with no spills).
// ---------------------------------------------------------------------------

typedef __attribute__((ext_vector_type(16))) _Float16 v16h;
typedef __attribute__((ext_vector_type(8)))  _Float16 v8h;
typedef __attribute__((ext_vector_type(4)))  _Float16 v4h;
typedef __attribute__((ext_vector_type(8)))  float    v8f;
typedef __attribute__((ext_vector_type(4)))  float    v4f;

#define CCH    64     // channels
#define LSEQ   8192   // sequence length
#define NBATCH 32
#define TILE   256    // output positions per block
#define WTILE  32     // output positions per wave (two N=16 WMMA tiles)
#define HALO   4      // max pad (k=9)
#define XPOS   (TILE + 2 * HALO)   // 264 positions resident in LDS
#define STR    72     // f16 per LDS position row (64 + pad, 144B bank skew)
#define SLICE  4096   // halves per 64x64 weight slice

union U16h { v16h v; v8h p[2]; };

// A fragment (16x32 f16, M x K): weight slice row-major [o][i].
// Lanes 0-15 -> row M=lane, halves 0..7 = K 0..7, 8..15 = K 16..23;
// lanes 16-31 -> row M=lane-16, halves 0..7 = K 8..15, 8..15 = K 24..31.
__device__ inline v16h load_a(const _Float16* __restrict__ wslice,
                              int mt, int cb, int lane) {
  const int row  = mt * 16 + (lane & 15);
  const int koff = (lane < 16) ? 0 : 8;
  const _Float16* p = wslice + row * CCH + cb * 32 + koff;
  U16h u;
  u.p[0] = *(const v8h*)(p);
  u.p[1] = *(const v8h*)(p + 16);
  return u.v;
}

// B fragment (32x16 f16, K x N) from LDS stored [pos][ch] (stride STR).
// Lanes 0-15: column N=lane, K 0..7,16..23; lanes 16-31: K 8..15,24..31.
__device__ inline v16h load_b(const _Float16* base, int posbase, int cb, int lane) {
  const int pos   = posbase + (lane & 15);
  const int kbase = cb * 32 + ((lane < 16) ? 0 : 8);
  const _Float16* q = base + pos * STR + kbase;
  U16h u;
  u.p[0] = *(const v8h*)(q);
  u.p[1] = *(const v8h*)(q + 16);
  return u.v;
}

// C/D fragment bias init: lane holds column n, VGPR r holds M = r + 8*(lane/16)
__device__ inline v8f bias_frag(const float* __restrict__ b, int mt, int lane) {
  const float* p = b + mt * 16 + ((lane < 16) ? 0 : 8);
  v4f a = *(const v4f*)p;
  v4f c = *(const v4f*)(p + 4);
  v8f r;
  r[0] = a[0]; r[1] = a[1]; r[2] = a[2]; r[3] = a[3];
  r[4] = c[0]; r[5] = c[1]; r[6] = c[2]; r[7] = c[3];
  return r;
}

// Prefetch ns weight slices (8KB each) into near caches: 2 lines/lane/slice.
__device__ inline void prefetch_slices(const _Float16* w, int ns, int lane) {
  for (int s = 0; s < ns; ++s) {
    __builtin_prefetch(w + s * SLICE + lane * 64, 0, 3);
    __builtin_prefetch(w + s * SLICE + (lane + 32) * 64, 0, 3);
  }
}

template <int K, int PAD>
__device__ inline void run_branch(const _Float16* xs, _Float16* hw,
                                  const _Float16* __restrict__ w1,
                                  const _Float16* __restrict__ w2,   // pre-scaled by sw_i
                                  const float* __restrict__ b1,
                                  const _Float16* wnext, int nnext,
                                  int posbase, int lane,
                                  v8f acc[4][2]) {
  // ---- first conv: h = b1 + sum_dk W1[dk] * X[.,pos+dk-PAD]
  v8f h[4][2];
#pragma unroll
  for (int mt = 0; mt < 4; ++mt) {
    v8f bfr = bias_frag(b1, mt, lane);
    h[mt][0] = bfr; h[mt][1] = bfr;
  }
#pragma unroll
  for (int dk = 0; dk < K; ++dk) {
    const _Float16* wslice = w1 + dk * SLICE;
#pragma unroll
    for (int cb = 0; cb < 2; ++cb) {
      v16h bf0 = load_b(xs, posbase + HALO + dk - PAD, cb, lane);
      v16h bf1 = load_b(xs, posbase + 16 + HALO + dk - PAD, cb, lane);
#pragma unroll
      for (int mt = 0; mt < 4; ++mt) {
        v16h af = load_a(wslice, mt, cb, lane);   // reused for both N-tiles
        h[mt][0] = __builtin_amdgcn_wmma_f32_16x16x32_f16(
            false, af, false, bf0, (short)0, h[mt][0], false, false);
        h[mt][1] = __builtin_amdgcn_wmma_f32_16x16x32_f16(
            false, af, false, bf1, (short)0, h[mt][1], false, false);
      }
    }
  }
  // overlap: pull next branch's weight slices toward WGP$
  prefetch_slices(wnext, nnext, lane);

  // ---- ReLU -> f16 -> per-wave LDS staging as [pos][ch]
  const int n  = lane & 15;
  const int hb = (lane >> 4) & 1;
#pragma unroll
  for (int nt = 0; nt < 2; ++nt) {
#pragma unroll
    for (int mt = 0; mt < 4; ++mt) {
      v8h hh;
#pragma unroll
      for (int r = 0; r < 8; ++r) {
        float v = h[mt][nt][r];
        hh[r] = (_Float16)(v > 0.f ? v : 0.f);
      }
      *(v8h*)(hw + (nt * 16 + n) * STR + mt * 16 + hb * 8) = hh;
    }
  }
  // ---- second conv (1x1, weights pre-scaled by softmax weight):
  //      accumulate straight into the persistent blended accumulator.
#pragma unroll
  for (int cb = 0; cb < 2; ++cb) {
    v16h bf0 = load_b(hw, 0,  cb, lane);
    v16h bf1 = load_b(hw, 16, cb, lane);
#pragma unroll
    for (int mt = 0; mt < 4; ++mt) {
      v16h af = load_a(w2, mt, cb, lane);
      acc[mt][0] = __builtin_amdgcn_wmma_f32_16x16x32_f16(
          false, af, false, bf0, (short)0, acc[mt][0], false, false);
      acc[mt][1] = __builtin_amdgcn_wmma_f32_16x16x32_f16(
          false, af, false, bf1, (short)0, acc[mt][1], false, false);
    }
  }
}

struct Params {
  const float* x;
  const _Float16* w;     // 24 f16 weight slices in workspace (w2 pre-scaled)
  const float* bb;       // blended bias sum_i sw_i * b2_i (64 floats, in ws)
  const float* b1[4];
  float* out;
};

__global__ __launch_bounds__(256, 2) void wavelet_fused_kernel(Params p) {
  __shared__ _Float16 xs[XPOS * STR];          // 38,016 B
  __shared__ _Float16 hs[8 * WTILE * STR];     // 36,864 B
  const int tid  = threadIdx.x;
  const int lane = tid & 31;
  const int wv   = tid >> 5;
  const int b    = blockIdx.x / (LSEQ / TILE);
  const int lts  = (blockIdx.x % (LSEQ / TILE)) * TILE;

  // warm the first branch's weights while x streams in
  prefetch_slices(p.w, 3, lane);
  prefetch_slices(p.w + 20 * SLICE, 1, lane);

  // ---- stage x tile (fp32 global -> f16 LDS), 4 channels per iteration
  const float* xb = p.x + (size_t)b * LSEQ * CCH;
  for (int t = tid; t < XPOS * (CCH / 4); t += 256) {
    const int hp = t >> 4;           // position 0..263
    const int c4 = (t & 15) * 4;     // channel chunk
    const int l  = lts - HALO + hp;
    v4f v = {0.f, 0.f, 0.f, 0.f};
    if (l >= 0 && l < LSEQ) v = *(const v4f*)(xb + (size_t)l * CCH + c4);
    v4h h4;
    h4[0] = (_Float16)v[0]; h4[1] = (_Float16)v[1];
    h4[2] = (_Float16)v[2]; h4[3] = (_Float16)v[3];
    *(v4h*)(xs + hp * STR + c4) = h4;
  }
  __syncthreads();

  // ---- persistent blended accumulator, seeded with sum_i sw_i*b2_i
  v8f acc[4][2];
#pragma unroll
  for (int mt = 0; mt < 4; ++mt) {
    v8f bfr = bias_frag(p.bb, mt, lane);
    acc[mt][0] = bfr; acc[mt][1] = bfr;
  }

  _Float16* hw = hs + wv * WTILE * STR;   // per-wave staging (32 positions)
  const int posbase = wv * WTILE;

  run_branch<3, 1>(xs, hw, p.w + 0  * SLICE, p.w + 20 * SLICE,
                   p.b1[0], p.w + 3  * SLICE, 4,  posbase, lane, acc);
  run_branch<3, 1>(xs, hw, p.w + 3  * SLICE, p.w + 21 * SLICE,
                   p.b1[1], p.w + 6  * SLICE, 6,  posbase, lane, acc);
  run_branch<5, 2>(xs, hw, p.w + 6  * SLICE, p.w + 22 * SLICE,
                   p.b1[2], p.w + 11 * SLICE, 10, posbase, lane, acc);
  run_branch<9, 4>(xs, hw, p.w + 11 * SLICE, p.w + 23 * SLICE,
                   p.b1[3], p.w + 20 * SLICE, 1,  posbase, lane, acc);

  // ---- writeback: lane owns positions posbase+{n, 16+n}, 8 ch / fragment
  const int n  = lane & 15;
  const int hb = (lane >> 4) & 1;
#pragma unroll
  for (int nt = 0; nt < 2; ++nt) {
    float* ob = p.out + ((size_t)(b * LSEQ + lts + posbase + nt * 16 + n)) * CCH;
#pragma unroll
    for (int mt = 0; mt < 4; ++mt) {
      float* o = ob + mt * 16 + hb * 8;
      v4f lo = {acc[mt][nt][0], acc[mt][nt][1], acc[mt][nt][2], acc[mt][nt][3]};
      v4f hi = {acc[mt][nt][4], acc[mt][nt][5], acc[mt][nt][6], acc[mt][nt][7]};
      *(v4f*)(o)     = lo;
      *(v4f*)(o + 4) = hi;
    }
  }
}

// ---- pre-kernels ----------------------------------------------------------

__device__ inline float softmax_w(const float* __restrict__ rec, int idx) {
  const float r0 = rec[0], r1 = rec[1], r2 = rec[2], r3 = rec[3];
  const float mx = fmaxf(fmaxf(r0, r1), fmaxf(r2, r3));
  const float e0 = __expf(r0 - mx), e1 = __expf(r1 - mx);
  const float e2 = __expf(r2 - mx), e3 = __expf(r3 - mx);
  const float e[4] = {e0, e1, e2, e3};
  return e[idx] / (e0 + e1 + e2 + e3);
}

// Convert one weight tensor [O=64][I=64][K] fp32 -> K slices of [O][I] f16.
__global__ void cvt_w1(const float* __restrict__ src,
                       _Float16* __restrict__ dst, int k, int total) {
  int i = blockIdx.x * 256 + threadIdx.x;
  if (i >= total) return;
  const int dk = i % k;
  const int ii = (i / k) % CCH;
  const int o  = i / (k * CCH);
  dst[dk * SLICE + o * CCH + ii] = (_Float16)src[i];
}

// Convert 1x1 weights [O][I][1] fp32 -> [O][I] f16 scaled by softmax weight.
__global__ void cvt_w2_scaled(const float* __restrict__ src,
                              const float* __restrict__ rec, int idx,
                              _Float16* __restrict__ dst) {
  int i = blockIdx.x * 256 + threadIdx.x;
  if (i >= CCH * CCH) return;
  dst[i] = (_Float16)(src[i] * softmax_w(rec, idx));
}

// Blend the four b2 biases with softmax weights -> 64 floats.
__global__ void blend_bias(const float* __restrict__ b0, const float* __restrict__ b1,
                           const float* __restrict__ b2, const float* __restrict__ b3,
                           const float* __restrict__ rec, float* __restrict__ dst) {
  int c = threadIdx.x;
  if (c >= CCH) return;
  dst[c] = softmax_w(rec, 0) * b0[c] + softmax_w(rec, 1) * b1[c] +
           softmax_w(rec, 2) * b2[c] + softmax_w(rec, 3) * b3[c];
}

extern "C" void kernel_launch(void* const* d_in, const int* in_sizes, int n_in,
                              void* d_out, int out_size, void* d_ws, size_t ws_size,
                              hipStream_t stream) {
  (void)in_sizes; (void)n_in; (void)out_size; (void)ws_size;
  // setup_inputs order:
  // 0:x 1:rec 2:trend_w1 3:trend_b1 4:trend_w2 5:trend_b2
  // 6:d1_w1 7:d1_b1 8:d1_w2 9:d1_b2  10:d2_w1 11:d2_b1 12:d2_w2 13:d2_b2
  // 14:d3_w1 15:d3_b1 16:d3_w2 17:d3_b2
  _Float16* wks = (_Float16*)d_ws;                 // 24*4096*2 = 196,608 B
  float*    bb  = (float*)(wks + 24 * SLICE);      // +256 B blended bias
  const float* rec = (const float*)d_in[1];

  struct Cv { const float* src; int k; int slice; };
  const Cv cv[4] = {
    { (const float*)d_in[2],  3,  0 },   // trend_w1
    { (const float*)d_in[6],  3,  3 },   // d1_w1
    { (const float*)d_in[10], 5,  6 },   // d2_w1
    { (const float*)d_in[14], 9, 11 },   // d3_w1
  };
  for (int i = 0; i < 4; ++i) {
    const int total = CCH * CCH * cv[i].k;
    cvt_w1<<<(total + 255) / 256, 256, 0, stream>>>(
        cv[i].src, wks + cv[i].slice * SLICE, cv[i].k, total);
  }
  const float* w2src[4] = {
    (const float*)d_in[4], (const float*)d_in[8],
    (const float*)d_in[12], (const float*)d_in[16],
  };
  for (int i = 0; i < 4; ++i) {
    cvt_w2_scaled<<<(CCH * CCH + 255) / 256, 256, 0, stream>>>(
        w2src[i], rec, i, wks + (20 + i) * SLICE);
  }
  blend_bias<<<1, 64, 0, stream>>>(
      (const float*)d_in[5], (const float*)d_in[9],
      (const float*)d_in[13], (const float*)d_in[17], rec, bb);

  Params p;
  p.x  = (const float*)d_in[0];
  p.w  = wks;
  p.bb = bb;
  p.b1[0] = (const float*)d_in[3];
  p.b1[1] = (const float*)d_in[7];
  p.b1[2] = (const float*)d_in[11];
  p.b1[3] = (const float*)d_in[15];
  p.out = (float*)d_out;

  const int grid = NBATCH * (LSEQ / TILE);   // 1024 blocks, 8 waves each
  wavelet_fused_kernel<<<grid, 256, 0, stream>>>(p);
}
